// BinLinear_29205777612937
// MI455X (gfx1250) — matrix operand order repaired and checked by reading, not compiled, not run on hardware
//
#include <hip/hip_runtime.h>
#include <hip/hip_bf16.h>

typedef __attribute__((ext_vector_type(16))) _Float16 v16h;
typedef __attribute__((ext_vector_type(8)))  _Float16 v8h;
typedef __attribute__((ext_vector_type(8)))  float    v8f;
typedef __attribute__((ext_vector_type(4)))  float    v4f;
typedef __attribute__((ext_vector_type(4)))  int      v4i;

#define AS1 __attribute__((address_space(1)))
#define AS3 __attribute__((address_space(3)))

#define N_ROWS 8192
#define D_IN   4096
#define D_OUT  4096

#define BM 128
#define BN 128
#define BK 64

#if __has_builtin(__builtin_amdgcn_global_load_async_to_lds_b128)
#define HAS_ASYNC_LDS 1
#define NBUF 2
#else
#define HAS_ASYNC_LDS 0
#define NBUF 1
#endif

__device__ __forceinline__ void wait_asynccnt0() {
#if __has_builtin(__builtin_amdgcn_s_wait_asynccnt)
    __builtin_amdgcn_s_wait_asynccnt(0);
#else
    asm volatile("s_wait_asynccnt 0x0" ::: "memory");
#endif
}

// ---------------- Pass 1a: x (f32) -> f16 ----------------
__global__ __launch_bounds__(256) void cvt_x_f16(const float* __restrict__ x,
                                                 _Float16* __restrict__ xh) {
    int i = (blockIdx.x * 256 + threadIdx.x) * 8;
    v4f a = *(const v4f*)(x + i);
    v4f b = *(const v4f*)(x + i + 4);
    v8h o;
#pragma unroll
    for (int j = 0; j < 4; ++j) o[j]     = (_Float16)a[j];
#pragma unroll
    for (int j = 0; j < 4; ++j) o[j + 4] = (_Float16)b[j];
    *(v8h*)(xh + i) = o;
}

// ---------------- Pass 1b: sign(W) (f32) -> f16 (exact {-1,0,1}) ----------------
__global__ __launch_bounds__(256) void cvt_w_sign(const float* __restrict__ w,
                                                  _Float16* __restrict__ wh) {
    int i = (blockIdx.x * 256 + threadIdx.x) * 8;
    v4f a = *(const v4f*)(w + i);
    v4f b = *(const v4f*)(w + i + 4);
    v8h o;
#pragma unroll
    for (int j = 0; j < 4; ++j)
        o[j]     = (_Float16)((a[j] > 0.f) ? 1.f : ((a[j] < 0.f) ? -1.f : 0.f));
#pragma unroll
    for (int j = 0; j < 4; ++j)
        o[j + 4] = (_Float16)((b[j] > 0.f) ? 1.f : ((b[j] < 0.f) ? -1.f : 0.f));
    *(v8h*)(wh + i) = o;
}

// ---------------- Pass 2: WMMA GEMM  out = Xh @ Wh^T + sign(bias) ----------------
// Xh: [N_ROWS][D_IN] f16 row-major (A, K-contiguous)
// Wh: [D_OUT][D_IN] f16 row-major  (column-major view of B, K-contiguous)
__global__ __launch_bounds__(256) void bin_gemm_wmma(
    const _Float16* __restrict__ Xh,
    const _Float16* __restrict__ Wh,
    const float*    __restrict__ bias,
    float*          __restrict__ out)
{
    __shared__ __align__(128) _Float16 As[NBUF][BM][BK];   // 16 KB per buf
    __shared__ __align__(128) _Float16 Bs[NBUF][BN][BK];   // 16 KB per buf

    const int tid  = threadIdx.x;
    const int lane = tid & 31;
    const int wave = tid >> 5;          // 0..7
    const int wm   = wave & 3;          // M offset = wm*32
    const int wn   = wave >> 2;         // N offset = wn*64
    const int lh   = lane & 15;         // lane within half
    const int hi16 = (lane & 16) ? 1 : 0;

    const int blockM = blockIdx.y * BM;
    const int blockN = blockIdx.x * BN;

    v8f acc[2][4];
#pragma unroll
    for (int tm = 0; tm < 2; ++tm)
#pragma unroll
        for (int tn = 0; tn < 4; ++tn)
            acc[tm][tn] = (v8f)0.f;

    // ---- compute: two K=32 WMMA steps against buffer `buf` ----
    auto compute = [&](int buf) {
#pragma unroll
        for (int ks = 0; ks < 2; ++ks) {
            // A fragments (16x32 f16): lane m holds row m.
            // lanes 0-15: K 0..7 & 16..23 ; lanes 16-31: K 8..15 & 24..31
            v16h af[2];
#pragma unroll
            for (int tm = 0; tm < 2; ++tm) {
                int row = wm * 32 + tm * 16 + lh;
                int k0  = ks * 32 + hi16 * 8;
                v8h lo = *(const v8h*)&As[buf][row][k0];
                v8h hp = *(const v8h*)&As[buf][row][k0 + 16];
#pragma unroll
                for (int j = 0; j < 8; ++j) { af[tm][j] = lo[j]; af[tm][j + 8] = hp[j]; }
            }
            // B fragments (32x16 f16): lane n holds column n.
            // lanes 0-15: K 0..7 & 16..23 ; lanes 16-31: K 8..15 & 24..31
            v16h bf[4];
#pragma unroll
            for (int tn = 0; tn < 4; ++tn) {
                int coln = wn * 64 + tn * 16 + lh;
                int k0   = ks * 32 + hi16 * 8;
                v8h blo = *(const v8h*)&Bs[buf][coln][k0];
                v8h bhp = *(const v8h*)&Bs[buf][coln][k0 + 16];
#pragma unroll
                for (int j = 0; j < 8; ++j) { bf[tn][j] = blo[j]; bf[tn][j + 8] = bhp[j]; }
            }
#pragma unroll
            for (int tm = 0; tm < 2; ++tm)
#pragma unroll
                for (int tn = 0; tn < 4; ++tn)
                    acc[tm][tn] = __builtin_amdgcn_wmma_f32_16x16x32_f16(
                        false, af[tm], false, bf[tn],
                        (short)0, acc[tm][tn], false, false);
        }
    };

#if HAS_ASYNC_LDS
    // ---- async global->LDS staging, no VGPR round-trip (ASYNCcnt tracked) ----
    auto stage_async = [&](int kb, int buf) {
#pragma unroll
        for (int i = 0; i < 4; ++i) {
            int c   = tid + i * 256;
            int r   = c >> 3;            // 0..127
            int col = (c & 7) * 8;       // 8 f16 = 16B per chunk
            __builtin_amdgcn_global_load_async_to_lds_b128(
                (v4i*)(Xh + (size_t)(blockM + r) * D_IN + kb + col),
                (AS3 v4i*)&As[buf][r][col], 0, 0);
            __builtin_amdgcn_global_load_async_to_lds_b128(
                (v4i*)(Wh + (size_t)(blockN + r) * D_IN + kb + col),
                (AS3 v4i*)&Bs[buf][r][col], 0, 0);
        }
    };

    stage_async(0, 0);
    wait_asynccnt0();
    __syncthreads();
    int cur = 0;
    for (int kb = 0; kb < D_IN; kb += BK) {
        if (kb + BK < D_IN) stage_async(kb + BK, cur ^ 1);  // prefetch next tile
        compute(cur);                                       // 16 WMMAs
        wait_asynccnt0();                                   // prefetch landed
        __syncthreads();                                    // all waves done reading cur
        cur ^= 1;
    }
#else
    // ---- fallback: synchronous staging through VGPRs, single buffer ----
    for (int kb = 0; kb < D_IN; kb += BK) {
#pragma unroll
        for (int i = 0; i < 4; ++i) {
            int c   = tid + i * 256;
            int r   = c >> 3;
            int col = (c & 7) * 8;
            *(v8h*)&As[0][r][col] =
                *(const v8h*)(Xh + (size_t)(blockM + r) * D_IN + kb + col);
            *(v8h*)&Bs[0][r][col] =
                *(const v8h*)(Wh + (size_t)(blockN + r) * D_IN + kb + col);
        }
        __syncthreads();
        compute(0);
        __syncthreads();
    }
#endif

    // ---- epilogue: C layout VGPR r -> M = r (+8 for lanes 16-31), N = lane&15 ----
#pragma unroll
    for (int tm = 0; tm < 2; ++tm) {
#pragma unroll
        for (int tn = 0; tn < 4; ++tn) {
            int n     = blockN + wn * 64 + tn * 16 + lh;
            int mBase = blockM + wm * 32 + tm * 16 + hi16 * 8;
            float bv  = bias[n];
            float bs  = (bv > 0.f) ? 1.f : ((bv < 0.f) ? -1.f : 0.f);
#pragma unroll
            for (int r = 0; r < 8; ++r)
                out[(size_t)(mBase + r) * D_OUT + n] = acc[tm][tn][r] + bs;
        }
    }
}

extern "C" void kernel_launch(void* const* d_in, const int* in_sizes, int n_in,
                              void* d_out, int out_size, void* d_ws, size_t ws_size,
                              hipStream_t stream) {
    const float* x      = (const float*)d_in[0];   // [8192,4096]
    const float* weight = (const float*)d_in[1];   // [4096,4096]
    const float* bias   = (const float*)d_in[2];   // [4096]
    float*       out    = (float*)d_out;           // [8192,4096]

    _Float16* xh = (_Float16*)d_ws;                                      // 64 MB
    _Float16* wh = (_Float16*)((char*)d_ws + (size_t)N_ROWS * D_IN * 2); // 32 MB

    // Pass 1: f32 -> f16 conversions (x) and exact sign binarization (W).
    cvt_x_f16 <<<(N_ROWS * (size_t)D_IN) / (256 * 8), 256, 0, stream>>>(x, xh);
    cvt_w_sign<<<(D_OUT  * (size_t)D_IN) / (256 * 8), 256, 0, stream>>>(weight, wh);

    // Pass 2: WMMA GEMM. 96 MB f16 working set is L2-resident (192 MB L2).
    dim3 grid(D_OUT / BN, N_ROWS / BM);
    bin_gemm_wmma<<<grid, 256, 0, stream>>>(xh, wh, bias, out);
}
